// GNN_TrackLinkingNet_55499567399311
// MI455X (gfx1250) — matrix operand order repaired and checked by reading, not compiled, not run on hardware
//
#include <hip/hip_runtime.h>
#include <hip/hip_bf16.h>
#include <math.h>

typedef __attribute__((ext_vector_type(16))) _Float16 v16h;
typedef __attribute__((ext_vector_type(8)))  float    v8f;

#define WPB 8          // waves per block (wave32)
#define BLOCK 256
#define CBAR() asm volatile("" ::: "memory")

// ---------------- WMMA helpers (gfx1250, wave32) ----------------

__device__ __forceinline__ v8f wmma_f16(v16h a, v16h b, v8f c) {
  // v_wmma_f32_16x16x32_f16: D = A(16x32 f16) * B(32x16 f16) + C(16x16 f32)
  return __builtin_amdgcn_wmma_f32_16x16x32_f16(false, a, false, b, (short)0, c, false, false);
}

__device__ __forceinline__ v8f zero8() {
  v8f z;
#pragma unroll
  for (int i = 0; i < 8; ++i) z[i] = 0.0f;
  return z;
}

__device__ __forceinline__ v8f leaky8(v8f x) {
  v8f r;
#pragma unroll
  for (int i = 0; i < 8; ++i) r[i] = x[i] > 0.0f ? x[i] : 0.01f * x[i];
  return r;
}
__device__ __forceinline__ float leaky1(float x) { return x > 0.0f ? x : 0.01f * x; }
__device__ __forceinline__ float sigmoidf(float x) { return 1.0f / (1.0f + __expf(-x)); }

// C/D layout: lane holds col n = lane%16, rows m = 8*(lane/16)+r in element r.
__device__ __forceinline__ v8f bias_frag(const float* b, int ncol, int lane) {
  float bv = b[ncol + (lane & 15)];
  v8f c;
#pragma unroll
  for (int i = 0; i < 8; ++i) c[i] = bv;
  return c;
}

// B fragment: 16 output cols [ncol..ncol+15] of row-major W[K][ldn], K<=32, zero-padded.
// Layout: lanes 0-15 -> K 0-7 & 16-23, lanes 16-31 -> K 8-15 & 24-31, col = lane%16.
__device__ __forceinline__ v16h load_bfrag(const float* W, int K, int ldn, int ncol, int lane) {
  int n = (lane & 15) + ncol;
  int kb = (lane >> 4) << 3;
  v16h b;
#pragma unroll
  for (int v = 0; v < 8; ++v) {
    int k0 = kb + 2 * v + (v >= 4 ? 8 : 0), k1 = k0 + 1;
    b[2 * v]     = (_Float16)((k0 < K) ? W[k0 * ldn + n] : 0.0f);
    b[2 * v + 1] = (_Float16)((k1 < K) ? W[k1 * ldn + n] : 0.0f);
  }
  return b;
}

// A fragment (16x32 f16) from LDS staging [16][ldw], K window [kbase..kbase+31].
// Layout: lane row M=lane%16; lanes 0-15: K 0-7 & 16-23, lanes 16-31: K 8-15 & 24-31.
__device__ __forceinline__ v16h load_afrag(const _Float16* s, int ldw, int kbase, int lane) {
  int m = lane & 15;
  int off = (lane >> 4) << 3;
  const _Float16* row = s + m * ldw + kbase + off;
  v16h a;
#pragma unroll
  for (int i = 0; i < 8; ++i) a[i] = row[i];
#pragma unroll
  for (int i = 0; i < 8; ++i) a[8 + i] = row[16 + i];
  return a;
}

// Store D fragment (f32) into LDS staging as f16 at cols [ncol..ncol+15].
__device__ __forceinline__ void store_dfrag_lds(v8f d, _Float16* s, int ldw, int ncol, int lane) {
  int n = (lane & 15) + ncol;
  int mb = (lane >> 4) << 3;
#pragma unroll
  for (int r = 0; r < 8; ++r) s[(mb + r) * ldw + n] = (_Float16)d[r];
}

// Sum across the 16 lanes of each wave half (cols of one D row block).
__device__ __forceinline__ void red16(float (&p)[8]) {
#pragma unroll
  for (int r = 0; r < 8; ++r) {
    float v = p[r];
    v += __shfl_xor(v, 1);
    v += __shfl_xor(v, 2);
    v += __shfl_xor(v, 4);
    v += __shfl_xor(v, 8);
    p[r] = v;
  }
}

// ---------------- node = mlp2(X / node_scaler, node_in) ----------------
__global__ void k_node_in(const float* __restrict__ X, const float* __restrict__ nsc,
                          const float* __restrict__ W1, const float* __restrict__ b1,
                          const float* __restrict__ W2, const float* __restrict__ b2,
                          float* __restrict__ node, int N) {
  __shared__ _Float16 stage[WPB][16][32];
  int lane = threadIdx.x & 31, wave = threadIdx.x >> 5;
  _Float16(*s)[32] = stage[wave];
  int T = (N + 15) >> 4;
  int nwaves = gridDim.x * WPB;
  v16h B1 = load_bfrag(W1, 19, 16, 0, lane);
  v16h B2 = load_bfrag(W2, 16, 16, 0, lane);
  v8f C1 = bias_frag(b1, 0, lane), C2 = bias_frag(b2, 0, lane);
  for (int tile = blockIdx.x * WPB + wave; tile < T; tile += nwaves) {
    int base = tile << 4;
    if (lane < 16) {
      int i = base + lane; if (i >= N) i = N - 1;
      const float* xr = X + (long)i * 19;
#pragma unroll
      for (int k = 0; k < 19; ++k) s[lane][k] = (_Float16)(xr[k] / nsc[k]);
#pragma unroll
      for (int k = 19; k < 32; ++k) s[lane][k] = (_Float16)0.0f;
    }
    CBAR();
    v16h A1 = load_afrag(&s[0][0], 32, 0, lane);
    v8f h1 = leaky8(wmma_f16(A1, B1, C1));
    CBAR();
    store_dfrag_lds(h1, &s[0][0], 32, 0, lane);
    store_dfrag_lds(zero8(), &s[0][0], 32, 16, lane);
    CBAR();
    v16h A2 = load_afrag(&s[0][0], 32, 0, lane);
    v8f h2 = leaky8(wmma_f16(A2, B2, C2));
    int n = lane & 15, mb = (lane >> 4) << 3;
#pragma unroll
    for (int r = 0; r < 8; ++r) {
      int i = base + mb + r;
      if (i < N) node[i * 16 + n] = h2[r];
    }
    CBAR();
  }
}

// ---------------- ef, efn = mlp2(ef, edge_in), alpha = [att_dir, att_rev] ----------------
__global__ void k_edge_in(const float* __restrict__ EFt, const float* __restrict__ esc,
                          const float* W1, const float* b1, const float* W2, const float* b2,
                          const float* Wd1, const float* bd1, const float* Wd2, const float* bd2,
                          const float* Wr1, const float* br1, const float* Wr2, const float* br2,
                          _Float16* __restrict__ efn, float* __restrict__ alpha, int E) {
  __shared__ _Float16 stage[WPB][16][32];
  int lane = threadIdx.x & 31, wave = threadIdx.x >> 5;
  _Float16(*s)[32] = stage[wave];
  int T = (E + 15) >> 4;
  int nwaves = gridDim.x * WPB;
  v16h B1 = load_bfrag(W1, 12, 16, 0, lane);
  v16h B2 = load_bfrag(W2, 16, 16, 0, lane);
  v16h BD = load_bfrag(Wd1, 16, 16, 0, lane);
  v16h BR = load_bfrag(Wr1, 16, 16, 0, lane);
  v8f C1 = bias_frag(b1, 0, lane), C2 = bias_frag(b2, 0, lane);
  v8f CD = bias_frag(bd1, 0, lane), CR = bias_frag(br1, 0, lane);
  float wd2 = Wd2[lane & 15], wr2 = Wr2[lane & 15];
  float bd2v = bd2[0], br2v = br2[0];
  for (int tile = blockIdx.x * WPB + wave; tile < T; tile += nwaves) {
    int base = tile << 4;
    if (lane < 16) {
      int e = base + lane; if (e >= E) e = E - 1;
      const float* fr = EFt + (long)e * 12;
#pragma unroll
      for (int k = 0; k < 12; ++k) s[lane][k] = (_Float16)((fr[k] + 1e-4f) / esc[k]);
#pragma unroll
      for (int k = 12; k < 32; ++k) s[lane][k] = (_Float16)0.0f;
    }
    CBAR();
    v16h A1 = load_afrag(&s[0][0], 32, 0, lane);
    v8f h1 = leaky8(wmma_f16(A1, B1, C1));
    CBAR();
    store_dfrag_lds(h1, &s[0][0], 32, 0, lane);  // cols 16-31 stay zero
    CBAR();
    v16h A2 = load_afrag(&s[0][0], 32, 0, lane);
    v8f ef2 = leaky8(wmma_f16(A2, B2, C2));
    int n = lane & 15, mb = (lane >> 4) << 3;
#pragma unroll
    for (int r = 0; r < 8; ++r) {
      int e = base + mb + r;
      if (e < E) efn[(long)e * 16 + n] = (_Float16)ef2[r];
    }
    CBAR();
    store_dfrag_lds(ef2, &s[0][0], 32, 0, lane);
    CBAR();
    v16h Ae = load_afrag(&s[0][0], 32, 0, lane);
    v8f hd = leaky8(wmma_f16(Ae, BD, CD));
    v8f hr = leaky8(wmma_f16(Ae, BR, CR));
    float pd[8], pr[8];
#pragma unroll
    for (int r = 0; r < 8; ++r) { pd[r] = hd[r] * wd2; pr[r] = hr[r] * wr2; }
    red16(pd);
    red16(pr);
    if ((lane & 15) == 0) {
#pragma unroll
      for (int r = 0; r < 8; ++r) {
        int e = base + mb + r;
        if (e < E) {
          alpha[e]     = sigmoidf(pd[r] + bd2v);
          alpha[E + e] = sigmoidf(pr[r] + br2v);
        }
      }
    }
    CBAR();
  }
}

// ---------------- conv: per-message MLP + weighted scatter-add ----------------
__global__ void k_msg(const float* __restrict__ node, const int* __restrict__ ei,
                      const float* __restrict__ alpha,
                      const float* W1, const float* b1, const float* W2, const float* b2,
                      float* __restrict__ num, float* __restrict__ den, int E) {
  __shared__ _Float16 stage[WPB][16][32];
  __shared__ float sw[WPB][16];
  __shared__ int sp[WPB][16];
  int lane = threadIdx.x & 31, wave = threadIdx.x >> 5;
  _Float16(*s)[32] = stage[wave];
  int M = 2 * E;
  int T = (M + 15) >> 4;
  int nwaves = gridDim.x * WPB;
  v16h B1a = load_bfrag(W1, 32, 32, 0, lane);
  v16h B1b = load_bfrag(W1, 32, 32, 16, lane);
  v16h B2  = load_bfrag(W2, 32, 16, 0, lane);
  v8f C1a = bias_frag(b1, 0, lane), C1b = bias_frag(b1, 16, lane), C2 = bias_frag(b2, 0, lane);
  for (int tile = blockIdx.x * WPB + wave; tile < T; tile += nwaves) {
    int base = tile << 4;
    int row = lane & 15, half = lane >> 4;
    int msg = base + row;
    bool valid = msg < M;
    int msgc = valid ? msg : M - 1;
    bool dir = msgc < E;
    int e = dir ? msgc : msgc - E;
    int si = ei[2 * e], di = ei[2 * e + 1];
    int p1 = dir ? si : di;
    int p2 = dir ? di : si;
    float w = alpha[msgc];
    if (half == 0) {
      sw[wave][row] = w;
      sp[wave][row] = p1;
      if (valid) atomicAdd(&den[p1], w);
    }
    // each lane loads 8 feats of xi and xj for its half of the row
    const float* xi = node + (long)p1 * 16 + half * 8;
    const float* xj = node + (long)p2 * 16 + half * 8;
#pragma unroll
    for (int k = 0; k < 8; ++k) {
      float a = xi[k], b = xj[k];
      s[row][half * 8 + k]      = (_Float16)a;        // m[:, 0:16]  = xi
      s[row][16 + half * 8 + k] = (_Float16)(b - a);  // m[:, 16:32] = xj - xi
    }
    CBAR();
    v16h A = load_afrag(&s[0][0], 32, 0, lane);
    v8f h1a = leaky8(wmma_f16(A, B1a, C1a));
    v8f h1b = leaky8(wmma_f16(A, B1b, C1b));
    CBAR();
    store_dfrag_lds(h1a, &s[0][0], 32, 0, lane);
    store_dfrag_lds(h1b, &s[0][0], 32, 16, lane);
    CBAR();
    v16h A2 = load_afrag(&s[0][0], 32, 0, lane);
    v8f h = leaky8(wmma_f16(A2, B2, C2));
    int n = lane & 15, mb = half << 3;
#pragma unroll
    for (int r = 0; r < 8; ++r) {
      int rr = mb + r;
      if (base + rr < M)
        atomicAdd(&num[(long)sp[wave][rr] * 16 + n], sw[wave][rr] * h[r]);
    }
    CBAR();
  }
}

// The first N messages all connect node 0 -> node 0 with w=1 (zeros placeholder in the
// reference). They are identical, so evaluate once and scale by N.
__global__ void k_self(const float* __restrict__ node,
                       const float* W1, const float* b1, const float* W2, const float* b2,
                       float* __restrict__ num, float* __restrict__ den, int N) {
  if (threadIdx.x == 0 && blockIdx.x == 0) {
    float x0[16];
    for (int k = 0; k < 16; ++k) x0[k] = node[k];
    float hid[32];
    for (int j = 0; j < 32; ++j) {
      float a = b1[j];
      for (int k = 0; k < 16; ++k) a += x0[k] * W1[k * 32 + j];  // m[16:32] == 0
      hid[j] = leaky1(a);
    }
    for (int n = 0; n < 16; ++n) {
      float a = b2[n];
      for (int j = 0; j < 32; ++j) a += hid[j] * W2[j * 16 + n];
      atomicAdd(&num[n], (float)N * leaky1(a));
    }
    atomicAdd(den, (float)N);
  }
}

__global__ void k_zero(float* __restrict__ p, long n) {
  long i = (long)blockIdx.x * blockDim.x + threadIdx.x;
  if (i < n) p[i] = 0.0f;
}

__global__ void k_div(float* __restrict__ num, const float* __restrict__ den, int N) {
  int i = blockIdx.x * blockDim.x + threadIdx.x;
  if (i < N * 16) num[i] = num[i] / fmaxf(den[i >> 4], 1e-8f);
}

// ---------------- edgenet + out head ----------------
__global__ void k_final(const float* __restrict__ node, const int* __restrict__ ei,
                        const _Float16* __restrict__ efn, const float* __restrict__ EFt,
                        const float* __restrict__ esc,
                        const float* W1, const float* b1, const float* W2, const float* b2,
                        const float* Wo1, const float* bo1, const float* Wo2, const float* bo2,
                        float* __restrict__ out, int E) {
  __shared__ _Float16 stage[WPB][16][64];
  int lane = threadIdx.x & 31, wave = threadIdx.x >> 5;
  _Float16(*s)[64] = stage[wave];
  int T = (E + 15) >> 4;
  int nwaves = gridDim.x * WPB;
  v16h BL = load_bfrag(W1, 32, 16, 0, lane);            // edgenet.W1 rows 0..31
  v16h BH = load_bfrag(W1 + 32 * 16, 28, 16, 0, lane);  // rows 32..59 (pad 60..63)
  v16h B2 = load_bfrag(W2, 16, 16, 0, lane);
  v16h BO = load_bfrag(Wo1, 16, 16, 0, lane);
  v8f C1 = bias_frag(b1, 0, lane), C2 = bias_frag(b2, 0, lane), CO = bias_frag(bo1, 0, lane);
  float wo2 = Wo2[lane & 15];
  float bo2v = bo2[0];
  for (int tile = blockIdx.x * WPB + wave; tile < T; tile += nwaves) {
    int base = tile << 4;
    if (lane < 16) {
      int e = base + lane; if (e >= E) e = E - 1;
      int si = ei[2 * e], di = ei[2 * e + 1];
      const float* ns = node + (long)si * 16;
      const float* nd = node + (long)di * 16;
#pragma unroll
      for (int k = 0; k < 16; ++k) s[lane][k] = (_Float16)ns[k];
#pragma unroll
      for (int k = 0; k < 16; ++k) s[lane][16 + k] = (_Float16)nd[k];
      const _Float16* er = efn + (long)e * 16;
#pragma unroll
      for (int k = 0; k < 16; ++k) s[lane][32 + k] = er[k];
      const float* fr = EFt + (long)e * 12;
#pragma unroll
      for (int k = 0; k < 12; ++k) s[lane][48 + k] = (_Float16)((fr[k] + 1e-4f) / esc[k]);
#pragma unroll
      for (int k = 60; k < 64; ++k) s[lane][k] = (_Float16)0.0f;
    }
    CBAR();
    v16h Alo = load_afrag(&s[0][0], 64, 0, lane);
    v16h Ahi = load_afrag(&s[0][0], 64, 32, lane);
    v8f acc = wmma_f16(Alo, BL, C1);
    acc = wmma_f16(Ahi, BH, acc);
    v8f h1 = leaky8(acc);
    CBAR();
    store_dfrag_lds(h1, &s[0][0], 64, 0, lane);
    store_dfrag_lds(zero8(), &s[0][0], 64, 16, lane);
    CBAR();
    v16h A2 = load_afrag(&s[0][0], 64, 0, lane);
    v8f emb = leaky8(wmma_f16(A2, B2, C2));
    CBAR();
    store_dfrag_lds(emb, &s[0][0], 64, 0, lane);  // cols 16-31 stay zero
    CBAR();
    v16h A3 = load_afrag(&s[0][0], 64, 0, lane);
    v8f ho = leaky8(wmma_f16(A3, BO, CO));
    float p[8];
#pragma unroll
    for (int r = 0; r < 8; ++r) p[r] = ho[r] * wo2;
    red16(p);
    int mb = (lane >> 4) << 3;
    if ((lane & 15) == 0) {
#pragma unroll
      for (int r = 0; r < 8; ++r) {
        int e = base + mb + r;
        if (e < E) out[e] = sigmoidf(p[r] + bo2v);
      }
    }
    CBAR();
  }
}

// ---------------- host launcher ----------------
extern "C" void kernel_launch(void* const* d_in, const int* in_sizes, int n_in,
                              void* d_out, int out_size, void* d_ws, size_t ws_size,
                              hipStream_t stream) {
  (void)in_sizes; (void)n_in; (void)out_size; (void)ws_size;
  const int N = 100000, E = 1600000;

  const float* X    = (const float*)d_in[0];
  const float* EFt  = (const float*)d_in[1];
  const int*   EI   = (const int*)d_in[2];
  const float* niW1 = (const float*)d_in[3];
  const float* nib1 = (const float*)d_in[4];
  const float* niW2 = (const float*)d_in[5];
  const float* nib2 = (const float*)d_in[6];
  const float* eiW1 = (const float*)d_in[7];
  const float* eib1 = (const float*)d_in[8];
  const float* eiW2 = (const float*)d_in[9];
  const float* eib2 = (const float*)d_in[10];
  const float* adW1 = (const float*)d_in[11];
  const float* adb1 = (const float*)d_in[12];
  const float* adW2 = (const float*)d_in[13];
  const float* adb2 = (const float*)d_in[14];
  const float* arW1 = (const float*)d_in[15];
  const float* arb1 = (const float*)d_in[16];
  const float* arW2 = (const float*)d_in[17];
  const float* arb2 = (const float*)d_in[18];
  const float* enW1 = (const float*)d_in[27];
  const float* enb1 = (const float*)d_in[28];
  const float* enW2 = (const float*)d_in[29];
  const float* enb2 = (const float*)d_in[30];
  const float* oW1  = (const float*)d_in[31];
  const float* ob1  = (const float*)d_in[32];
  const float* oW2  = (const float*)d_in[33];
  const float* ob2  = (const float*)d_in[34];
  const float* nsc  = (const float*)d_in[35];
  const float* esc  = (const float*)d_in[36];

  // workspace carve-out (~70 MB)
  char* ws = (char*)d_ws;
  size_t off = 0;
  auto carve = [&](size_t bytes) { void* p = ws + off; off = (off + bytes + 255) & ~(size_t)255; return p; };
  _Float16* efn = (_Float16*)carve((size_t)E * 16 * sizeof(_Float16));
  float* alpha  = (float*)carve((size_t)2 * E * sizeof(float));
  float* nodeA  = (float*)carve((size_t)N * 16 * sizeof(float));
  float* nodeB  = (float*)carve((size_t)N * 16 * sizeof(float));
  float* den    = (float*)carve((size_t)N * sizeof(float));

  {
    int T = (N + 15) / 16;
    int blocks = (T + WPB - 1) / WPB;
    k_node_in<<<blocks, BLOCK, 0, stream>>>(X, nsc, niW1, nib1, niW2, nib2, nodeA, N);
  }
  k_edge_in<<<8192, BLOCK, 0, stream>>>(EFt, esc, eiW1, eib1, eiW2, eib2,
                                        adW1, adb1, adW2, adb2,
                                        arW1, arb1, arW2, arb2, efn, alpha, E);

  float* cur = nodeA;
  float* nxt = nodeB;
  for (int it = 0; it < 2; ++it) {
    const float* cW1 = (const float*)d_in[19 + 4 * it];
    const float* cb1 = (const float*)d_in[20 + 4 * it];
    const float* cW2 = (const float*)d_in[21 + 4 * it];
    const float* cb2 = (const float*)d_in[22 + 4 * it];
    long nn = (long)N * 16;
    k_zero<<<(int)((nn + BLOCK - 1) / BLOCK), BLOCK, 0, stream>>>(nxt, nn);
    k_zero<<<(N + BLOCK - 1) / BLOCK, BLOCK, 0, stream>>>(den, (long)N);
    k_self<<<1, 32, 0, stream>>>(cur, cW1, cb1, cW2, cb2, nxt, den, N);
    k_msg<<<8192, BLOCK, 0, stream>>>(cur, EI, alpha, cW1, cb1, cW2, cb2, nxt, den, E);
    k_div<<<(int)((nn + BLOCK - 1) / BLOCK), BLOCK, 0, stream>>>(nxt, den, N);
    float* t = cur; cur = nxt; nxt = t;
  }

  k_final<<<8192, BLOCK, 0, stream>>>(cur, EI, efn, EFt, esc,
                                      enW1, enb1, enW2, enb2,
                                      oW1, ob1, oW2, ob2, (float*)d_out, E);
}
